// CustomMambaDualScan_74620761801335
// MI455X (gfx1250) — compile-verified, matrix operand order checked by reading, not run on hardware
//
#include <hip/hip_runtime.h>
#include <hip/hip_bf16.h>

typedef __bf16 bf16_t;
typedef __attribute__((ext_vector_type(16))) __bf16 v16bf;
typedef __attribute__((ext_vector_type(8)))  float  v8f;

// Problem constants
constexpr int BD   = 2;           // batch
constexpr int TS   = 128;         // time steps
constexpr int NH   = 64;          // heads / Nn
constexpr int FD   = 256;         // feature F
constexpr int RD   = 128;         // RED
constexpr int ED   = 256;         // E = 2*RED
constexpr int NST  = 16;          // SSM states
constexpr int DBLW = 40;          // R + 2*NS = 8 + 32
constexpr int NSEQ = BD * NH;     // 128 sequences
constexpr int ROWS = NSEQ * TS;   // 16384 (seq-major rows)

static __device__ __forceinline__ float siluf(float x) {
    return x / (1.0f + __expf(-x));
}

// ---------------------------------------------------------------------------
// Weight prep: transpose f32 (rows x cols) -> bf16 (cols x rows)
// so WMMA B-fragments read contiguous K per output column.
// ---------------------------------------------------------------------------
__global__ void k_transpose_bf16(const float* __restrict__ src,
                                 bf16_t* __restrict__ dst, int rows, int cols) {
    int i = blockIdx.x * 256 + threadIdx.x;
    if (i < rows * cols) {
        int r = i / cols, c = i - r * cols;
        dst[(size_t)c * rows + r] = (bf16_t)src[i];
    }
}

// ===========================================================================
// WMMA GEMM waves: each wave computes one 16-row M tile x four 16-col N tiles
// (A fragment loaded/converted once per K-step, reused for 4 WMMAs).
// Fragment layouts per CDNA5 ISA:
//   A: lane&15 = row, lane>>4 = K-half; elems 0..7 -> K=kh*8+i, 8..15 -> +16
//   B: lane&15 = col, same K mapping (B stored transposed: [n][k])
//   C/D: elem j -> M = j + 8*(lane>>4), N = lane&15
// ===========================================================================

// ---------------------------------------------------------------------------
// K1: h = concat(x, qk) @ W_in + b_in     (M=16384, K=512, N=128)
// Output written seq-major: h[(seq*TS + t)*RD + n],  seq = b*NH + nh.
// ---------------------------------------------------------------------------
__global__ __launch_bounds__(256) void k_gemm_in(
    const float* __restrict__ x, const float* __restrict__ qk,
    const bf16_t* __restrict__ WinT, const float* __restrict__ b_in,
    float* __restrict__ h) {
    constexpr int KD   = 2 * FD;     // 512
    constexpr int NGRP = RD / 64;    // 2 groups of 4 N-tiles
    int wave = threadIdx.x >> 5, lane = threadIdx.x & 31;
    int task = blockIdx.x * 8 + wave;
    int tM = task / NGRP, tG = task - tM * NGRP;
    int m0 = tM * 16, n0 = tG * 64;
    int lr = lane & 15, kh = lane >> 4;
    int mg = m0 + lr;
    const float* ax = x  + (size_t)mg * FD;
    const float* aq = qk + (size_t)mg * FD;
    const bf16_t* br[4];
#pragma unroll
    for (int tn = 0; tn < 4; ++tn)
        br[tn] = WinT + (size_t)(n0 + tn * 16 + lr) * KD;
    v8f acc[4] = {{}, {}, {}, {}};
    for (int kb = 0; kb < KD; kb += 32) {
        v16bf a;
        int ka = kb + kh * 8;
#pragma unroll
        for (int i = 0; i < 8; ++i) {
            int k0 = ka + i, k1 = ka + 16 + i;
            a[i]     = (bf16_t)((k0 < FD) ? ax[k0] : aq[k0 - FD]);
            a[8 + i] = (bf16_t)((k1 < FD) ? ax[k1] : aq[k1 - FD]);
        }
#pragma unroll
        for (int tn = 0; tn < 4; ++tn) {
            v16bf b;
#pragma unroll
            for (int i = 0; i < 8; ++i) {
                b[i]     = br[tn][ka + i];
                b[8 + i] = br[tn][ka + 16 + i];
            }
            acc[tn] = __builtin_amdgcn_wmma_f32_16x16x32_bf16(
                false, a, false, b, (short)0, acc[tn], false, false);
        }
    }
#pragma unroll
    for (int tn = 0; tn < 4; ++tn) {
        int ng = n0 + tn * 16 + lr;
        float bias = b_in[ng];
#pragma unroll
        for (int j = 0; j < 8; ++j) {
            int mm  = m0 + j + 8 * kh;         // global A row = ((b*TS+t)*NH+nh)
            int bi  = mm / (TS * NH);
            int rem = mm - bi * (TS * NH);
            int t   = rem / NH;
            int nh  = rem - t * NH;
            int seq = bi * NH + nh;
            h[((size_t)seq * TS + t) * RD + ng] = acc[tn][j] + bias;
        }
    }
}

// ---------------------------------------------------------------------------
// K2: xz = u @ in_proj   (M=16384, K=128, N=512), time-flip baked into A read.
// Splits columns into xi_pre (0..255) and z (256..511), stored in scan order.
// ---------------------------------------------------------------------------
__global__ __launch_bounds__(256) void k_gemm_xz(
    const float* __restrict__ h, const bf16_t* __restrict__ ipT,
    float* __restrict__ xi_pre, float* __restrict__ z, int flip) {
    constexpr int KD   = RD;          // 128
    constexpr int ND   = 2 * ED;      // 512
    constexpr int NGRP = ND / 64;     // 8
    int wave = threadIdx.x >> 5, lane = threadIdx.x & 31;
    int task = blockIdx.x * 8 + wave;
    int tM = task / NGRP, tG = task - tM * NGRP;
    int m0 = tM * 16, n0 = tG * 64;
    int lr = lane & 15, kh = lane >> 4;
    int mg  = m0 + lr;
    int seq = mg / TS, t = mg - seq * TS;
    int ts  = flip ? (TS - 1 - t) : t;
    const float* arow = h + ((size_t)seq * TS + ts) * RD;
    const bf16_t* br[4];
#pragma unroll
    for (int tn = 0; tn < 4; ++tn)
        br[tn] = ipT + (size_t)(n0 + tn * 16 + lr) * KD;
    v8f acc[4] = {{}, {}, {}, {}};
    for (int kb = 0; kb < KD; kb += 32) {
        v16bf a;
        int ka = kb + kh * 8;
#pragma unroll
        for (int i = 0; i < 8; ++i) {
            a[i]     = (bf16_t)arow[ka + i];
            a[8 + i] = (bf16_t)arow[ka + 16 + i];
        }
#pragma unroll
        for (int tn = 0; tn < 4; ++tn) {
            v16bf b;
#pragma unroll
            for (int i = 0; i < 8; ++i) {
                b[i]     = br[tn][ka + i];
                b[8 + i] = br[tn][ka + 16 + i];
            }
            acc[tn] = __builtin_amdgcn_wmma_f32_16x16x32_bf16(
                false, a, false, b, (short)0, acc[tn], false, false);
        }
    }
#pragma unroll
    for (int tn = 0; tn < 4; ++tn) {
        int nn = n0 + tn * 16 + lr;
#pragma unroll
        for (int j = 0; j < 8; ++j) {
            int mm = m0 + j + 8 * kh;
            if (nn < ED) xi_pre[(size_t)mm * ED + nn]      = acc[tn][j];
            else         z     [(size_t)mm * ED + nn - ED] = acc[tn][j];
        }
    }
}

// ---------------------------------------------------------------------------
// K3a: causal depthwise conv (K=4) + bias + SiLU
// ---------------------------------------------------------------------------
__global__ void k_conv_silu(const float* __restrict__ xi_pre,
                            const float* __restrict__ cw,
                            const float* __restrict__ cb,
                            float* __restrict__ xi_post) {
    int idx = blockIdx.x * 256 + threadIdx.x;
    if (idx >= ROWS * ED) return;
    int e   = idx & (ED - 1);
    int row = idx >> 8;               // ED == 256
    int t   = row & (TS - 1);
    float acc = cb[e];
#pragma unroll
    for (int k = 0; k < 4; ++k) {
        int tt = t - 3 + k;
        if (tt >= 0) acc += xi_pre[(size_t)(row - 3 + k) * ED + e] * cw[e * 4 + k];
    }
    xi_post[idx] = siluf(acc);
}

// ---------------------------------------------------------------------------
// K3b: dbl = xi_post @ x_proj   (256 -> 40)
// ---------------------------------------------------------------------------
__global__ void k_dbl(const float* __restrict__ xi_post,
                      const float* __restrict__ xproj, float* __restrict__ dbl) {
    int idx = blockIdx.x * 256 + threadIdx.x;
    if (idx >= ROWS * DBLW) return;
    int row = idx / DBLW, c = idx - row * DBLW;
    const float* xr = xi_post + (size_t)row * ED;
    float s = 0.f;
    for (int k = 0; k < ED; ++k) s += xr[k] * xproj[k * DBLW + c];
    dbl[idx] = s;
}

// ---------------------------------------------------------------------------
// K3c: dt = softplus(dbl[:, :8] @ dt_w + dt_b)   (8 -> 256)
// ---------------------------------------------------------------------------
__global__ void k_dt(const float* __restrict__ dbl,
                     const float* __restrict__ dtw,
                     const float* __restrict__ dtb, float* __restrict__ dt) {
    int idx = blockIdx.x * 256 + threadIdx.x;
    if (idx >= ROWS * ED) return;
    int e = idx & (ED - 1);
    int row = idx >> 8;
    float s = dtb[e];
#pragma unroll
    for (int r = 0; r < 8; ++r) s += dbl[(size_t)row * DBLW + r] * dtw[r * ED + e];
    dt[idx] = (s > 20.f) ? s : log1pf(__expf(s));
}

// ---------------------------------------------------------------------------
// K4: selective scan. One block per sequence; thread e owns h[e][0..15] in
// registers; B_t / C_t broadcast through LDS. Fuses D-skip and SiLU(z) gate.
// ---------------------------------------------------------------------------
__global__ __launch_bounds__(256) void k_scan(
    const float* __restrict__ dt, const float* __restrict__ dbl,
    const float* __restrict__ xi_post, const float* __restrict__ z,
    const float* __restrict__ A_log, const float* __restrict__ Dp,
    float* __restrict__ yg) {
    int seq = blockIdx.x;
    int e   = threadIdx.x;
    float a[NST], hst[NST];
#pragma unroll
    for (int n = 0; n < NST; ++n) {
        a[n]   = -__expf(A_log[e * NST + n]);
        hst[n] = 0.f;
    }
    float dskip = Dp[e];
    __shared__ float Bs_[NST], Cs_[NST];
    for (int t = 0; t < TS; ++t) {
        size_t row = (size_t)seq * TS + t;
        if (threadIdx.x < 32) {
            float v = dbl[row * DBLW + 8 + threadIdx.x];
            if (threadIdx.x < 16) Bs_[threadIdx.x] = v;
            else                  Cs_[threadIdx.x - 16] = v;
        }
        __syncthreads();
        float dtv = dt[row * ED + e];
        float xv  = xi_post[row * ED + e];
        float dx  = dtv * xv;
        float y   = 0.f;
#pragma unroll
        for (int n = 0; n < NST; ++n) {
            float w = __expf(dtv * a[n]);
            hst[n]  = w * hst[n] + dx * Bs_[n];
            y      += hst[n] * Cs_[n];
        }
        float zv = z[row * ED + e];
        yg[row * ED + e] = (y + xv * dskip) * siluf(zv);
        __syncthreads();
    }
}

// ---------------------------------------------------------------------------
// K5: ycomb (+)= yg @ out_proj   (M=16384, K=256, N=128); un-flips bw time.
// ---------------------------------------------------------------------------
__global__ __launch_bounds__(256) void k_gemm_out(
    const float* __restrict__ yg, const bf16_t* __restrict__ opT,
    float* __restrict__ ycomb, int flip, int accum) {
    constexpr int KD   = ED;          // 256
    constexpr int NGRP = RD / 64;     // 2
    int wave = threadIdx.x >> 5, lane = threadIdx.x & 31;
    int task = blockIdx.x * 8 + wave;
    int tM = task / NGRP, tG = task - tM * NGRP;
    int m0 = tM * 16, n0 = tG * 64;
    int lr = lane & 15, kh = lane >> 4;
    const float* arow = yg + (size_t)(m0 + lr) * KD;
    const bf16_t* br[4];
#pragma unroll
    for (int tn = 0; tn < 4; ++tn)
        br[tn] = opT + (size_t)(n0 + tn * 16 + lr) * KD;
    v8f acc[4] = {{}, {}, {}, {}};
    for (int kb = 0; kb < KD; kb += 32) {
        v16bf a;
        int ka = kb + kh * 8;
#pragma unroll
        for (int i = 0; i < 8; ++i) {
            a[i]     = (bf16_t)arow[ka + i];
            a[8 + i] = (bf16_t)arow[ka + 16 + i];
        }
#pragma unroll
        for (int tn = 0; tn < 4; ++tn) {
            v16bf b;
#pragma unroll
            for (int i = 0; i < 8; ++i) {
                b[i]     = br[tn][ka + i];
                b[8 + i] = br[tn][ka + 16 + i];
            }
            acc[tn] = __builtin_amdgcn_wmma_f32_16x16x32_bf16(
                false, a, false, b, (short)0, acc[tn], false, false);
        }
    }
#pragma unroll
    for (int tn = 0; tn < 4; ++tn) {
        int nn = n0 + tn * 16 + lr;
#pragma unroll
        for (int j = 0; j < 8; ++j) {
            int mm   = m0 + j + 8 * kh;
            int seq  = mm / TS, t = mm - seq * TS;
            int tout = flip ? (TS - 1 - t) : t;
            size_t oi = ((size_t)seq * TS + tout) * RD + nn;
            if (accum) ycomb[oi] += acc[tn][j];
            else       ycomb[oi]  = acc[tn][j];
        }
    }
}

// ---------------------------------------------------------------------------
// K6a: LayerNorm over RD=128, one wave per row (4 elems/lane + shfl reduce)
// ---------------------------------------------------------------------------
__global__ __launch_bounds__(256) void k_ln(
    const float* __restrict__ ycomb, const float* __restrict__ ln_g,
    const float* __restrict__ ln_b, float* __restrict__ ynorm) {
    int row  = blockIdx.x * 8 + (threadIdx.x >> 5);
    int lane = threadIdx.x & 31;
    const float* src = ycomb + (size_t)row * RD;
    float v[4], s = 0.f, s2 = 0.f;
#pragma unroll
    for (int i = 0; i < 4; ++i) {
        v[i] = src[lane + 32 * i];
        s  += v[i];
        s2 += v[i] * v[i];
    }
    for (int off = 16; off >= 1; off >>= 1) {
        s  += __shfl_xor(s,  off, 32);
        s2 += __shfl_xor(s2, off, 32);
    }
    float mu   = s * (1.0f / RD);
    float var  = s2 * (1.0f / RD) - mu * mu;
    float rinv = rsqrtf(var + 1e-5f);
#pragma unroll
    for (int i = 0; i < 4; ++i) {
        int c = lane + 32 * i;
        ynorm[(size_t)row * RD + c] = (v[i] - mu) * rinv * ln_g[c] + ln_b[c];
    }
}

// ---------------------------------------------------------------------------
// K6b: out = ynorm @ W_out + b_out  (M=16384, K=128, N=256); seq-major ->
// (b, t, nh, f) layout on store.
// ---------------------------------------------------------------------------
__global__ __launch_bounds__(256) void k_gemm_final(
    const float* __restrict__ ynorm, const bf16_t* __restrict__ woT,
    const float* __restrict__ b_out, float* __restrict__ out) {
    constexpr int KD   = RD;          // 128
    constexpr int NGRP = FD / 64;     // 4
    int wave = threadIdx.x >> 5, lane = threadIdx.x & 31;
    int task = blockIdx.x * 8 + wave;
    int tM = task / NGRP, tG = task - tM * NGRP;
    int m0 = tM * 16, n0 = tG * 64;
    int lr = lane & 15, kh = lane >> 4;
    const float* arow = ynorm + (size_t)(m0 + lr) * KD;
    const bf16_t* br[4];
#pragma unroll
    for (int tn = 0; tn < 4; ++tn)
        br[tn] = woT + (size_t)(n0 + tn * 16 + lr) * KD;
    v8f acc[4] = {{}, {}, {}, {}};
    for (int kb = 0; kb < KD; kb += 32) {
        v16bf a;
        int ka = kb + kh * 8;
#pragma unroll
        for (int i = 0; i < 8; ++i) {
            a[i]     = (bf16_t)arow[ka + i];
            a[8 + i] = (bf16_t)arow[ka + 16 + i];
        }
#pragma unroll
        for (int tn = 0; tn < 4; ++tn) {
            v16bf b;
#pragma unroll
            for (int i = 0; i < 8; ++i) {
                b[i]     = br[tn][ka + i];
                b[8 + i] = br[tn][ka + 16 + i];
            }
            acc[tn] = __builtin_amdgcn_wmma_f32_16x16x32_bf16(
                false, a, false, b, (short)0, acc[tn], false, false);
        }
    }
#pragma unroll
    for (int tn = 0; tn < 4; ++tn) {
        int nn = n0 + tn * 16 + lr;
        float bias = b_out[nn];
#pragma unroll
        for (int j = 0; j < 8; ++j) {
            int mm  = m0 + j + 8 * kh;     // seq-major row
            int seq = mm / TS, t = mm - seq * TS;
            int bi  = seq / NH, nh = seq - bi * NH;
            out[(((size_t)bi * TS + t) * NH + nh) * FD + nn] = acc[tn][j] + bias;
        }
    }
}

// ---------------------------------------------------------------------------
extern "C" void kernel_launch(void* const* d_in, const int* in_sizes, int n_in,
                              void* d_out, int out_size, void* d_ws, size_t ws_size,
                              hipStream_t stream) {
    (void)in_sizes; (void)n_in; (void)out_size; (void)ws_size;

    const float* x     = (const float*)d_in[0];
    const float* qk    = (const float*)d_in[1];
    const float* W_in  = (const float*)d_in[2];
    const float* b_in  = (const float*)d_in[3];
    const float* ln_g  = (const float*)d_in[4];
    const float* ln_b  = (const float*)d_in[5];
    const float* W_out = (const float*)d_in[6];
    const float* b_out = (const float*)d_in[7];
    // fw params: 8..16, bw params: 17..25 (in_proj, conv_w, conv_b, x_proj,
    // dt_w, dt_b, A_log, D, out_proj)
    const float* m_in_proj[2]  = {(const float*)d_in[8],  (const float*)d_in[17]};
    const float* m_conv_w[2]   = {(const float*)d_in[9],  (const float*)d_in[18]};
    const float* m_conv_b[2]   = {(const float*)d_in[10], (const float*)d_in[19]};
    const float* m_x_proj[2]   = {(const float*)d_in[11], (const float*)d_in[20]};
    const float* m_dt_w[2]     = {(const float*)d_in[12], (const float*)d_in[21]};
    const float* m_dt_b[2]     = {(const float*)d_in[13], (const float*)d_in[22]};
    const float* m_A_log[2]    = {(const float*)d_in[14], (const float*)d_in[23]};
    const float* m_D[2]        = {(const float*)d_in[15], (const float*)d_in[24]};
    const float* m_out_proj[2] = {(const float*)d_in[16], (const float*)d_in[25]};

    // Workspace carve-up
    size_t cur = 0;
    char* base = (char*)d_ws;
    auto allocF = [&](size_t n) { float*  p = (float*) (base + cur); cur += n * sizeof(float);  return p; };
    auto allocB = [&](size_t n) { bf16_t* p = (bf16_t*)(base + cur); cur += n * sizeof(bf16_t); cur = (cur + 15) & ~(size_t)15; return p; };

    float* h       = allocF((size_t)ROWS * RD);     //  8 MB
    float* xi_pre  = allocF((size_t)ROWS * ED);     // 16 MB
    float* zbuf    = allocF((size_t)ROWS * ED);     // 16 MB
    float* xi_post = allocF((size_t)ROWS * ED);     // 16 MB
    float* dblbuf  = allocF((size_t)ROWS * DBLW);   // 2.6 MB
    float* dtbuf   = allocF((size_t)ROWS * ED);     // 16 MB
    float* yg      = allocF((size_t)ROWS * ED);     // 16 MB
    float* ycomb   = allocF((size_t)ROWS * RD);     //  8 MB
    float* ynorm   = allocF((size_t)ROWS * RD);     //  8 MB
    bf16_t* wtin   = allocB(512 * 128);             // W_in^T      (N=128, K=512)
    bf16_t* ipT[2] = {allocB(512 * 128), allocB(512 * 128)};  // in_proj^T  (N=512, K=128)
    bf16_t* opT[2] = {allocB(256 * 128), allocB(256 * 128)};  // out_proj^T (N=128, K=256)
    bf16_t* woT    = allocB(256 * 128);             // W_out^T     (N=256, K=128)

    dim3 blk(256);
    auto tgrid = [](int n) { return dim3((n + 255) / 256); };

    // Weight prep (transpose + bf16)
    k_transpose_bf16<<<tgrid(512 * 128), blk, 0, stream>>>(W_in, wtin, 512, 128);
    k_transpose_bf16<<<tgrid(128 * 512), blk, 0, stream>>>(m_in_proj[0],  ipT[0], 128, 512);
    k_transpose_bf16<<<tgrid(128 * 512), blk, 0, stream>>>(m_in_proj[1],  ipT[1], 128, 512);
    k_transpose_bf16<<<tgrid(256 * 128), blk, 0, stream>>>(m_out_proj[0], opT[0], 256, 128);
    k_transpose_bf16<<<tgrid(256 * 128), blk, 0, stream>>>(m_out_proj[1], opT[1], 256, 128);
    k_transpose_bf16<<<tgrid(128 * 256), blk, 0, stream>>>(W_out, woT, 128, 256);

    // K1: input projection into seq-major h
    // tasks = (M/16)*(N/64) = 1024*2 = 2048 -> 256 blocks of 8 waves
    k_gemm_in<<<dim3(256), blk, 0, stream>>>(x, qk, wtin, b_in, h);

    // Two Mamba passes (fw = 0, bw = 1 with baked-in time flip)
    for (int dir = 0; dir < 2; ++dir) {
        // tasks = 1024*8 = 8192 -> 1024 blocks
        k_gemm_xz<<<dim3(1024), blk, 0, stream>>>(h, ipT[dir], xi_pre, zbuf, dir);
        k_conv_silu<<<tgrid(ROWS * ED), blk, 0, stream>>>(xi_pre, m_conv_w[dir],
                                                          m_conv_b[dir], xi_post);
        k_dbl<<<tgrid(ROWS * DBLW), blk, 0, stream>>>(xi_post, m_x_proj[dir], dblbuf);
        k_dt<<<tgrid(ROWS * ED), blk, 0, stream>>>(dblbuf, m_dt_w[dir],
                                                   m_dt_b[dir], dtbuf);
        k_scan<<<dim3(NSEQ), blk, 0, stream>>>(dtbuf, dblbuf, xi_post, zbuf,
                                               m_A_log[dir], m_D[dir], yg);
        // tasks = 1024*2 = 2048 -> 256 blocks
        k_gemm_out<<<dim3(256), blk, 0, stream>>>(yg, opT[dir], ycomb, dir, dir);
    }

    // LayerNorm + final projection to (b, t, nh, f)
    k_ln<<<dim3(2048), blk, 0, stream>>>(ycomb, ln_g, ln_b, ynorm);
    // tasks = 1024*4 = 4096 -> 512 blocks
    k_gemm_final<<<dim3(512), blk, 0, stream>>>(ynorm, woT, b_out, (float*)d_out);
}
